// DividedAttention_5660766896468
// MI455X (gfx1250) — compile-verified, matrix operand order checked by reading, not run on hardware
//
#include <hip/hip_runtime.h>

// ---------------------------------------------------------------- types ----
typedef _Float16 half_t;
typedef __attribute__((ext_vector_type(16))) _Float16 v16h;
typedef __attribute__((ext_vector_type(8)))  _Float16 v8h;
typedef __attribute__((ext_vector_type(4)))  _Float16 v4h;
typedef __attribute__((ext_vector_type(8)))  float    v8f;
typedef __attribute__((ext_vector_type(4)))  unsigned u32x4;
typedef __attribute__((ext_vector_type(8)))  int      i32x8;
typedef __attribute__((ext_vector_type(4)))  int      i32x4;

union V16H { v16h v; v8h h[2]; };

#define BATCH 8
#define HEADS 16
#define DH    64
#define DIM   1024
#define NSEQ  3137
#define FRAMES 16
#define NTOK  196      // tokens per frame
#define NKEY  197      // cls + tokens
#define NKP   208      // padded keys for score tiles (13*16)
#define NKPV  224      // padded keys for PV K-loop (7*32)

#if defined(__has_builtin)
#if __has_builtin(__builtin_amdgcn_tensor_load_to_lds)
#define HAVE_TDM 1
#endif
#endif
#ifndef HAVE_TDM
#define HAVE_TDM 0
#endif

// D = A(16x32 f16) * B(32x16 f16) + C(16x16 f32)
__device__ __forceinline__ v8f wmma16(v16h a, v16h b, v8f c) {
  return __builtin_amdgcn_wmma_f32_16x16x32_f16(false, a, false, b, (short)0, c,
                                                false, false);
}

// Operand fragment (A-style row / B-style column) from row-major [.][stride]
// half array. Per ISA: lane<16 holds K={0..7,16..23}, lane>=16 K={8..15,24..31}.
__device__ __forceinline__ v16h ld_frag(const half_t* __restrict__ base,
                                        int stride, int row, int kbase, int lane) {
  const half_t* p = base + (size_t)row * stride + kbase + ((lane >> 4) << 3);
  V16H u;
  u.h[0] = *(const v8h*)(p);
  u.h[1] = *(const v8h*)(p + 16);
  return u.v;
}

#if HAVE_TDM
// Issue a TDM load of a [rows=128] x [cols=32] f16 tile into LDS.
// 2D descriptor: tensor_dim1 = rowsRem so OOB rows read as zero (M edge tile).
__device__ __forceinline__ void tdm_load_tile_f16(const half_t* gsrc,
                                                  unsigned ldsBytes,
                                                  unsigned rowsRem) {
  unsigned long long ga = (unsigned long long)(const char*)gsrc;
  u32x4 g0;
  g0[0] = 1u;                                            // count=1, user mode
  g0[1] = ldsBytes;                                      // lds_addr
  g0[2] = (unsigned)(ga & 0xFFFFFFFFull);                // global_addr[31:0]
  g0[3] = (unsigned)((ga >> 32) & 0x1FFFFFFull) | (2u << 30); // addr[56:32]|type=2
  i32x8 g1;
  g1[0] = 0x10000;                  // workgroup_mask=0, data_size=1 (2 bytes)
  g1[1] = (int)(DIM << 16);         // tensor_dim0[15:0]=1024 (atomic_addr=0)
  g1[2] = (int)((rowsRem & 0xFFFFu) << 16);  // dim0 hi=0 | tensor_dim1 lo
  g1[3] = (int)(((rowsRem >> 16) & 0xFFFFu) | (32u << 16)); // dim1 hi | tile_dim0=32
  g1[4] = 128;                      // tile_dim1=128, tile_dim2=0
  g1[5] = DIM;                      // tensor_dim0_stride = 1024 elements
  g1[6] = 0;
  g1[7] = 0;
  i32x4 z4 = {};
#if __clang_major__ >= 23
  i32x8 z8 = {};
  __builtin_amdgcn_tensor_load_to_lds(g0, g1, z4, z4, z8, 0);
#else
  __builtin_amdgcn_tensor_load_to_lds(g0, g1, z4, z4, 0);
#endif
}
__device__ __forceinline__ unsigned lds_byte_addr(const void* p) {
  return (unsigned)(size_t)(__attribute__((address_space(3))) const char*)p;
}
#endif

// ============================================================ QKV GEMM =====
// C[m,n] = sum_k X[m,k] * W[n,k];  M=25096, K=1024, N=3072.
// Block tile 128x128, 8 waves (2x4), wave tile 64x32, double-buffered LDS.
// Output scattered as f16 into head-major q/k/v (b*h, NSEQ, 64); q scaled 1/8.
__global__ __launch_bounds__(256) void qkv_gemm(const float* __restrict__ X,
                                                const float* __restrict__ W,
                                                half_t* __restrict__ qh,
                                                half_t* __restrict__ kh,
                                                half_t* __restrict__ vh,
                                                int M) {
  __shared__ half_t As[2][128][32];
  __shared__ half_t Bs[2][128][32];
  const int tid  = threadIdx.x;
  const int lane = tid & 31;
  const int wid  = tid >> 5;
  const int wm   = wid >> 2;          // 0..1  (64 rows)
  const int wn   = wid & 3;           // 0..3  (32 cols)
  const int tileM = blockIdx.y * 128;
  const int tileN = blockIdx.x * 128;

  v8f acc[4][2] = {};

  float4 ar[4], br[4];
  auto fetch = [&](int k0) {
#pragma unroll
    for (int i = 0; i < 4; ++i) {
      int idx = tid + i * 256;        // 0..1023
      int r   = idx >> 3;
      int c4  = (idx & 7) * 4;
      int gr  = tileM + r;
      gr = (gr < M) ? gr : (M - 1);
      ar[i] = *(const float4*)(X + (size_t)gr * DIM + k0 + c4);
      br[i] = *(const float4*)(W + (size_t)(tileN + r) * DIM + k0 + c4);
    }
  };
  auto stage = [&](int b) {
#pragma unroll
    for (int i = 0; i < 4; ++i) {
      int idx = tid + i * 256;
      int r   = idx >> 3;
      int c4  = (idx & 7) * 4;
      v4h ha = { (half_t)ar[i].x, (half_t)ar[i].y, (half_t)ar[i].z, (half_t)ar[i].w };
      v4h hb = { (half_t)br[i].x, (half_t)br[i].y, (half_t)br[i].z, (half_t)br[i].w };
      *(v4h*)&As[b][r][c4] = ha;
      *(v4h*)&Bs[b][r][c4] = hb;
    }
  };

  fetch(0);
  stage(0);
  __syncthreads();

  int buf = 0;
  for (int k0 = 0; k0 < DIM; k0 += 32) {
    const bool more = (k0 + 32 < DIM);
    if (more) fetch(k0 + 32);         // global loads in flight over the WMMAs

    v16h bfrag[2];
#pragma unroll
    for (int s = 0; s < 2; ++s)
      bfrag[s] = ld_frag(&Bs[buf][0][0], 32, wn * 32 + s * 16 + (lane & 15), 0, lane);
#pragma unroll
    for (int t = 0; t < 4; ++t) {
      v16h af = ld_frag(&As[buf][0][0], 32, wm * 64 + t * 16 + (lane & 15), 0, lane);
#pragma unroll
      for (int s = 0; s < 2; ++s) acc[t][s] = wmma16(af, bfrag[s], acc[t][s]);
    }

    if (more) {
      stage(buf ^ 1);                 // disjoint buffer: one barrier suffices
      __syncthreads();
      buf ^= 1;
    }
  }

  // --- epilogue: scatter into q/k/v head-major f16
  const int which = tileN / DIM;      // uniform per block (128 | 1024)
  half_t* dst; float scale;
  if (which == 0)      { dst = qh; scale = 0.125f; }   // 1/sqrt(64)
  else if (which == 1) { dst = kh; scale = 1.0f; }
  else                 { dst = vh; scale = 1.0f; }

#pragma unroll
  for (int s = 0; s < 2; ++s) {
    int col  = tileN + wn * 32 + s * 16 + (lane & 15);
    int d    = col & (DIM - 1);
    int head = d >> 6;
    int wd   = d & 63;
#pragma unroll
    for (int t = 0; t < 4; ++t) {
#pragma unroll
      for (int r = 0; r < 8; ++r) {
        int m = tileM + wm * 64 + t * 16 + r + 8 * (lane >> 4);
        if (m < M) {
          int b   = m / NSEQ;
          int seq = m % NSEQ;
          size_t off = (((size_t)(b * HEADS + head)) * NSEQ + seq) * DH + wd;
          dst[off] = (half_t)(acc[t][s][r] * scale);
        }
      }
    }
  }
}

// ====================================================== divided attention ==
__global__ __launch_bounds__(128) void div_attn(const half_t* __restrict__ qh,
                                                const half_t* __restrict__ kh,
                                                const half_t* __restrict__ vh,
                                                half_t* __restrict__ attn) {
  __shared__ half_t Ks[NKP][DH];        // [key][dh]
  __shared__ half_t Vt[DH][NKPV];       // [dh][key] transposed
  __shared__ half_t Ps[4][16][NKPV];    // per-wave P restage

  const int g    = blockIdx.x;
  const int bh   = g >> 4;
  const int fr   = g & 15;
  const int tid  = threadIdx.x;
  const int lane = tid & 31;
  const int wid  = tid >> 5;
  const int head = bh & 15;
  const int bidx = bh >> 4;
  const size_t baseSeq = (size_t)bh * NSEQ;

  for (int idx = tid; idx < NKP * 8; idx += 128) {
    int r  = idx >> 3;
    int c8 = (idx & 7) * 8;
    v8h kv = {};
    if (r < NKEY) {
      int seq = (r == 0) ? 0 : (1 + fr * NTOK + (r - 1));
      kv = *(const v8h*)(kh + (baseSeq + seq) * DH + c8);
    }
    *(v8h*)&Ks[r][c8] = kv;
  }
  for (int idx = tid; idx < NKPV * 8; idx += 128) {
    int r  = idx >> 3;
    int c8 = (idx & 7) * 8;
    v8h vv = {};
    if (r < NKEY) {
      int seq = (r == 0) ? 0 : (1 + fr * NTOK + (r - 1));
      vv = *(const v8h*)(vh + (baseSeq + seq) * DH + c8);
    }
#pragma unroll
    for (int j = 0; j < 8; ++j) Vt[c8 + j][r] = vv[j];
  }
  __syncthreads();

  half_t* Pw = &Ps[wid][0][0];

  for (int qt = wid; qt < 13; qt += 4) {
    int rowIn = lane & 15;
    int qi    = qt * 16 + rowIn;
    int qic   = (qi < NTOK) ? qi : (NTOK - 1);
    const half_t* qp = qh + (baseSeq + 1 + (size_t)fr * NTOK + qic) * DH;
    int chunk = (lane >> 4) << 3;
    V16H q0, q1;
    q0.h[0] = *(const v8h*)(qp + chunk);
    q0.h[1] = *(const v8h*)(qp + chunk + 16);
    q1.h[0] = *(const v8h*)(qp + 32 + chunk);
    q1.h[1] = *(const v8h*)(qp + 32 + chunk + 16);

    v8f S[13];
#pragma unroll
    for (int kt = 0; kt < 13; ++kt) {
      v8f s = {};
      s = wmma16(q0.v, ld_frag(&Ks[0][0], DH, kt * 16 + (lane & 15), 0,  lane), s);
      s = wmma16(q1.v, ld_frag(&Ks[0][0], DH, kt * 16 + (lane & 15), 32, lane), s);
      S[kt] = s;
    }
    if ((lane & 15) >= 5) {
#pragma unroll
      for (int r = 0; r < 8; ++r) S[12][r] = -1e30f;
    }

    float lrow[8];
#pragma unroll
    for (int r = 0; r < 8; ++r) {
      float m = S[0][r];
#pragma unroll
      for (int kt = 1; kt < 13; ++kt) m = fmaxf(m, S[kt][r]);
      m = fmaxf(m, __shfl_xor(m, 1, 32));
      m = fmaxf(m, __shfl_xor(m, 2, 32));
      m = fmaxf(m, __shfl_xor(m, 4, 32));
      m = fmaxf(m, __shfl_xor(m, 8, 32));
      float sum = 0.f;
#pragma unroll
      for (int kt = 0; kt < 13; ++kt) {
        float p = __expf(S[kt][r] - m);
        S[kt][r] = p;
        sum += p;
      }
      sum += __shfl_xor(sum, 1, 32);
      sum += __shfl_xor(sum, 2, 32);
      sum += __shfl_xor(sum, 4, 32);
      sum += __shfl_xor(sum, 8, 32);
      lrow[r] = sum;
    }

#pragma unroll
    for (int kt = 0; kt < 13; ++kt)
#pragma unroll
      for (int r = 0; r < 8; ++r) {
        int row = r + 8 * (lane >> 4);
        int col = kt * 16 + (lane & 15);
        Pw[row * NKPV + col] = (half_t)S[kt][r];
      }
#pragma unroll
    for (int r = 0; r < 8; ++r)
      Pw[(r + 8 * (lane >> 4)) * NKPV + 208 + (lane & 15)] = (half_t)0.0f;

    v8f O[4] = {};
#pragma unroll
    for (int kk = 0; kk < 7; ++kk) {
      v16h pf = ld_frag(Pw, NKPV, lane & 15, kk * 32, lane);
#pragma unroll
      for (int nt = 0; nt < 4; ++nt) {
        v16h vf = ld_frag(&Vt[0][0], NKPV, nt * 16 + (lane & 15), kk * 32, lane);
        O[nt] = wmma16(pf, vf, O[nt]);
      }
    }

#pragma unroll
    for (int r = 0; r < 8; ++r) {
      int row = qt * 16 + r + 8 * (lane >> 4);
      if (row < NTOK) {
        int seq = 1 + fr * NTOK + row;
        float inv = 1.0f / lrow[r];
#pragma unroll
        for (int nt = 0; nt < 4; ++nt) {
          int col = head * DH + nt * 16 + (lane & 15);
          attn[((size_t)bidx * NSEQ + seq) * DIM + col] = (half_t)(O[nt][r] * inv);
        }
      }
    }
  }
}

// ========================================================= CLS attention ===
__global__ __launch_bounds__(128) void cls_attn(const half_t* __restrict__ qh,
                                                const half_t* __restrict__ kh,
                                                const half_t* __restrict__ vh,
                                                half_t* __restrict__ attn) {
  const int bh   = blockIdx.x * 4 + (threadIdx.x >> 5);
  const int lane = threadIdx.x & 31;
  const int bidx = bh >> 4;
  const int head = bh & 15;
  const size_t baseSeq = (size_t)bh * NSEQ;

  float q[DH];
  const half_t* qp = qh + baseSeq * DH;
#pragma unroll
  for (int d2 = 0; d2 < DH; ++d2) q[d2] = (float)qp[d2];

  float m = -1e30f, l = 0.f;
  float acc[DH];
#pragma unroll
  for (int d2 = 0; d2 < DH; ++d2) acc[d2] = 0.f;

  for (int j = lane; j < NSEQ; j += 32) {
    const half_t* kp = kh + (baseSeq + j) * DH;
    float s = 0.f;
#pragma unroll
    for (int d2 = 0; d2 < DH; ++d2) s += q[d2] * (float)kp[d2];
    float nm   = fmaxf(m, s);
    float corr = __expf(m - nm);
    float p    = __expf(s - nm);
    l = l * corr + p;
    const half_t* vp = vh + (baseSeq + j) * DH;
#pragma unroll
    for (int d2 = 0; d2 < DH; ++d2) acc[d2] = acc[d2] * corr + p * (float)vp[d2];
    m = nm;
  }
#pragma unroll
  for (int mask = 16; mask >= 1; mask >>= 1) {
    float om = __shfl_xor(m, mask, 32);
    float ol = __shfl_xor(l, mask, 32);
    float nm = fmaxf(m, om);
    float c1 = __expf(m - nm);
    float c2 = __expf(om - nm);
    l = l * c1 + ol * c2;
#pragma unroll
    for (int d2 = 0; d2 < DH; ++d2) {
      float oa = __shfl_xor(acc[d2], mask, 32);
      acc[d2] = acc[d2] * c1 + oa * c2;
    }
    m = nm;
  }
  if (lane == 0) {
    float inv = 1.0f / l;
    half_t* dst = attn + (size_t)bidx * NSEQ * DIM + head * DH;
#pragma unroll
    for (int d2 = 0; d2 < DH; ++d2) dst[d2] = (half_t)(acc[d2] * inv);
  }
}

// ============================================================ proj GEMM ====
// Out[m,n] = sum_k attn[m,k] * Wp[n,k] + bias[n].
// A tile (f16, pure copy) staged by the Tensor Data Mover when available.
__global__ __launch_bounds__(256) void proj_gemm(const half_t* __restrict__ A,
                                                 const float* __restrict__ Wp,
                                                 const float* __restrict__ bias,
                                                 float* __restrict__ Out,
                                                 int M) {
  __shared__ half_t As[2][128][32];
  __shared__ half_t Bs[2][128][32];
  const int tid  = threadIdx.x;
  const int lane = tid & 31;
  const int wid  = tid >> 5;
  const int wm   = wid >> 2;
  const int wn   = wid & 3;
  const int tileM = blockIdx.y * 128;
  const int tileN = blockIdx.x * 128;
  const unsigned rowsRem = (unsigned)(M - tileM);

  v8f acc[4][2] = {};

  float4 br[4];
  auto fetchB = [&](int k0) {
#pragma unroll
    for (int i = 0; i < 4; ++i) {
      int idx = tid + i * 256;
      int r   = idx >> 3;
      int c4  = (idx & 7) * 4;
      br[i] = *(const float4*)(Wp + (size_t)(tileN + r) * DIM + k0 + c4);
    }
  };
  auto stageB = [&](int b) {
#pragma unroll
    for (int i = 0; i < 4; ++i) {
      int idx = tid + i * 256;
      int r   = idx >> 3;
      int c4  = (idx & 7) * 4;
      v4h hb = { (half_t)br[i].x, (half_t)br[i].y, (half_t)br[i].z, (half_t)br[i].w };
      *(v4h*)&Bs[b][r][c4] = hb;
    }
  };
#if !HAVE_TDM
  v8h arh[2];
  auto fetchA = [&](int k0) {
#pragma unroll
    for (int i = 0; i < 2; ++i) {
      int idx = tid + i * 256;        // 0..511
      int r   = idx >> 2;
      int c8  = (idx & 3) * 8;
      int gr  = tileM + r;
      gr = (gr < M) ? gr : (M - 1);
      arh[i] = *(const v8h*)(A + (size_t)gr * DIM + k0 + c8);
    }
  };
  auto stageA = [&](int b) {
#pragma unroll
    for (int i = 0; i < 2; ++i) {
      int idx = tid + i * 256;
      int r   = idx >> 2;
      int c8  = (idx & 3) * 8;
      *(v8h*)&As[b][r][c8] = arh[i];
    }
  };
#endif

  // ---- preload k0 = 0
#if HAVE_TDM
  if (wid == 0)
    tdm_load_tile_f16(A + (size_t)tileM * DIM, lds_byte_addr(&As[0][0][0]), rowsRem);
#else
  fetchA(0);
#endif
  fetchB(0);
  stageB(0);
#if !HAVE_TDM
  stageA(0);
#endif
#if HAVE_TDM
  if (wid == 0) __builtin_amdgcn_s_wait_tensorcnt(0);
#endif
  __syncthreads();

  int buf = 0;
  for (int k0 = 0; k0 < DIM; k0 += 32) {
    const bool more = (k0 + 32 < DIM);
    if (more) {
#if HAVE_TDM
      if (wid == 0)
        tdm_load_tile_f16(A + (size_t)tileM * DIM + (k0 + 32),
                          lds_byte_addr(&As[buf ^ 1][0][0]), rowsRem);
#else
      fetchA(k0 + 32);
#endif
      fetchB(k0 + 32);
    }

    v16h bfrag[2];
#pragma unroll
    for (int s = 0; s < 2; ++s)
      bfrag[s] = ld_frag(&Bs[buf][0][0], 32, wn * 32 + s * 16 + (lane & 15), 0, lane);
#pragma unroll
    for (int t = 0; t < 4; ++t) {
      v16h af = ld_frag(&As[buf][0][0], 32, wm * 64 + t * 16 + (lane & 15), 0, lane);
#pragma unroll
      for (int s = 0; s < 2; ++s) acc[t][s] = wmma16(af, bfrag[s], acc[t][s]);
    }

    if (more) {
      stageB(buf ^ 1);
#if !HAVE_TDM
      stageA(buf ^ 1);
#endif
#if HAVE_TDM
      if (wid == 0) __builtin_amdgcn_s_wait_tensorcnt(0);
#endif
      __syncthreads();
      buf ^= 1;
    }
  }

#pragma unroll
  for (int s = 0; s < 2; ++s) {
    int col = tileN + wn * 32 + s * 16 + (lane & 15);
    float bv = bias[col];
#pragma unroll
    for (int t = 0; t < 4; ++t) {
#pragma unroll
      for (int r = 0; r < 8; ++r) {
        int m = tileM + wm * 64 + t * 16 + r + 8 * (lane >> 4);
        if (m < M) Out[(size_t)m * DIM + col] = acc[t][s][r] + bv;
      }
    }
  }
}

// ============================================================== launch =====
extern "C" void kernel_launch(void* const* d_in, const int* in_sizes, int n_in,
                              void* d_out, int out_size, void* d_ws, size_t ws_size,
                              hipStream_t stream) {
  const float* x      = (const float*)d_in[0];
  const float* qkv_w  = (const float*)d_in[1];
  const float* proj_w = (const float*)d_in[2];
  const float* proj_b = (const float*)d_in[3];
  float* out = (float*)d_out;

  const size_t HEADSZ = (size_t)BATCH * HEADS * NSEQ * DH;
  half_t* ws   = (half_t*)d_ws;
  half_t* qh   = ws;
  half_t* kh   = ws + HEADSZ;
  half_t* vh   = ws + 2 * HEADSZ;
  half_t* attn = ws + 3 * HEADSZ;

  const int M = BATCH * NSEQ;                               // 25096

  dim3 g1(3072 / 128, (M + 127) / 128);                     // 24 x 197
  qkv_gemm<<<g1, 256, 0, stream>>>(x, qkv_w, qh, kh, vh, M);

  div_attn<<<BATCH * HEADS * FRAMES, 128, 0, stream>>>(qh, kh, vh, attn);
  cls_attn<<<(BATCH * HEADS) / 4, 128, 0, stream>>>(qh, kh, vh, attn);

  dim3 g4(DIM / 128, (M + 127) / 128);                      // 8 x 197
  proj_gemm<<<g4, 256, 0, stream>>>(attn, proj_w, proj_b, out, M);
}
